// AtlasAttention_36094905156285
// MI455X (gfx1250) — matrix operand order repaired and checked by reading, not compiled, not run on hardware
//
#include <hip/hip_runtime.h>
#include <hip/hip_bf16.h>

// ---------------------------------------------------------------------------
// AtlasAttention fused pipeline for gfx1250 (MI455X), bf16 WMMA path.
//   prep : fp32 -> bf16 conversions, weight transposes, constant-fold bias
//   gemm1: Q = X @ Wq (8192x768x768) + poly epilogue -> poly[98304,192] bf16
//   mlp  : H = relu(poly @ W1nc + bias1) fused with out = H @ W2[:, :64] + b2
// ---------------------------------------------------------------------------

typedef __bf16 bf16x16 __attribute__((ext_vector_type(16)));
typedef float  f32x8   __attribute__((ext_vector_type(8)));

union Frag {
    bf16x16 v;
    uint4   u[2];
};

__device__ __forceinline__ unsigned short f2bf(float f) {
    unsigned int u = __float_as_uint(f);
    u += 0x7fffu + ((u >> 16) & 1u);     // round-to-nearest-even
    return (unsigned short)(u >> 16);
}

// ---- async direct-to-LDS staging (gfx1250), guarded with safe fallback ----
#if defined(__has_builtin)
#if __has_builtin(__builtin_amdgcn_global_load_async_to_lds_b128) && \
    __has_builtin(__builtin_amdgcn_global_load_async_to_lds_b64)  && \
    __has_builtin(__builtin_amdgcn_s_wait_asynccnt)
#define USE_ASYNC_LDS 1
#endif
#endif

#ifdef USE_ASYNC_LDS
typedef int v4i __attribute__((vector_size(16)));
typedef int v2i __attribute__((vector_size(8)));
// params are typed vector pointers: (addrspace(1) vNi*, addrspace(3) vNi*, imm, imm)
#define AS1_V4(p) ((__attribute__((address_space(1))) v4i*) \
                   (__attribute__((address_space(1))) void*)(void*)(p))
#define AS1_V2(p) ((__attribute__((address_space(1))) v2i*) \
                   (__attribute__((address_space(1))) void*)(void*)(p))
#define AS3_V4(p) ((__attribute__((address_space(3))) v4i*) \
                   (__attribute__((address_space(3))) void*)(p))
#define AS3_V2(p) ((__attribute__((address_space(3))) v2i*) \
                   (__attribute__((address_space(3))) void*)(p))
__device__ __forceinline__ void cp_async_b128(const void* g, void* l) {
    __builtin_amdgcn_global_load_async_to_lds_b128(AS1_V4(g), AS3_V4(l), 0, 0);
}
__device__ __forceinline__ void cp_async_b64(const void* g, void* l) {
    __builtin_amdgcn_global_load_async_to_lds_b64(AS1_V2(g), AS3_V2(l), 0, 0);
}
__device__ __forceinline__ void cp_async_wait() {
    __builtin_amdgcn_s_wait_asynccnt(0);
}
#endif

// ---------------- prep kernels ----------------

__global__ __launch_bounds__(256) void k_cvt_x(const float* __restrict__ x,
                                               unsigned short* __restrict__ xb, int n) {
    int i = (blockIdx.x * 256 + threadIdx.x) * 4;
    if (i + 3 < n) {
        float4 f = *(const float4*)(x + i);
        uint2 o;
        o.x = (unsigned)f2bf(f.x) | ((unsigned)f2bf(f.y) << 16);
        o.y = (unsigned)f2bf(f.z) | ((unsigned)f2bf(f.w) << 16);
        *(uint2*)(xb + i) = o;
    }
}

// Wq (in,out)[768][768] -> Wq_t[out][in] bf16 (N-major for B fragments)
__global__ __launch_bounds__(256) void k_wq_t(const float* __restrict__ wq,
                                              unsigned short* __restrict__ wqt) {
    int idx = blockIdx.x * 256 + threadIdx.x;           // 589824
    int o = idx / 768, i = idx - o * 768;
    wqt[idx] = f2bf(wq[i * 768 + o]);
}

// W1[64+k][n] -> W1nc_t[n][k] bf16  (n<512, k<192)
__global__ __launch_bounds__(256) void k_w1_t(const float* __restrict__ w1,
                                              unsigned short* __restrict__ w1t) {
    int idx = blockIdx.x * 256 + threadIdx.x;           // 98304
    int n = idx / 192, k = idx - n * 192;
    w1t[idx] = f2bf(w1[(64 + k) * 512 + n]);
}

// bias1f[n] = b1[n] + c0 * sum_{j<64} W1[j][n]  (fold constant poly feature)
__global__ __launch_bounds__(256) void k_bias1(const float* __restrict__ w1,
                                               const float* __restrict__ b1,
                                               const float* __restrict__ coeffs,
                                               float* __restrict__ bias1) {
    int n = blockIdx.x * 256 + threadIdx.x;             // 512
    if (n < 512) {
        float c0 = coeffs[0];
        float s = b1[n];
        for (int j = 0; j < 64; ++j) s += c0 * w1[j * 512 + n];
        bias1[n] = s;
    }
}

// W2[k][n] -> W2s_t[n][k] bf16  (n<64, k<512) — only surviving 64 columns
__global__ __launch_bounds__(256) void k_w2_t(const float* __restrict__ w2,
                                              unsigned short* __restrict__ w2t) {
    int idx = blockIdx.x * 256 + threadIdx.x;           // 32768
    int n = idx / 512, k = idx - n * 512;
    w2t[idx] = f2bf(w2[k * 256 + n]);
}

// ---------------- GEMM1: Q = X @ Wq, poly-feature epilogue ----------------
// block = 128 threads (4 waves), tile 64(M) x 64(N), K-step 32.
// NOTE: each N-block of 64 columns == exactly one head -> contiguous poly rows.
__global__ __launch_bounds__(128) void k_gemm1_poly(
    const unsigned short* __restrict__ Xb,   // [8192,768] bf16
    const unsigned short* __restrict__ Wt,   // [768,768]  bf16, N-major
    const float* __restrict__ coeffs,
    unsigned short* __restrict__ poly)       // [98304,192] bf16
{
    __shared__ __align__(16) unsigned short sA[64 * 32];
    __shared__ __align__(16) unsigned short sB[64 * 32];
    __shared__ __align__(16) unsigned short sP[64 * 192];   // 24 KB repack tile

    const int tid   = threadIdx.x;
    const int lane  = tid & 31;
    const int wave  = tid >> 5;              // 0..3
    const int waveM = (wave & 1) * 32;
    const int waveN = (wave >> 1) * 32;
    const int bM    = blockIdx.x * 64;
    const int bN    = blockIdx.y * 64;
    const int lrow  = lane & 15;
    const int kh    = lane >> 4;             // K-half select

    f32x8 acc[2][2] = {};

    const int row  = tid >> 1;               // 0..63 (tile staging)
    const int half = tid & 1;                // 32B chunk

    for (int kk = 0; kk < 768; kk += 32) {
        // stage A/B tiles into LDS: 32 bytes per thread each
        {
            const unsigned short* srcA = Xb + (bM + row) * 768 + kk + half * 16;
            unsigned short*       dstA = sA + row * 32 + half * 16;
            const unsigned short* srcB = Wt + (bN + row) * 768 + kk + half * 16;
            unsigned short*       dstB = sB + row * 32 + half * 16;
#ifdef USE_ASYNC_LDS
            cp_async_b128(srcA,     dstA);
            cp_async_b128(srcA + 8, dstA + 8);
            cp_async_b128(srcB,     dstB);
            cp_async_b128(srcB + 8, dstB + 8);
            cp_async_wait();
#else
            ((uint4*)dstA)[0] = ((const uint4*)srcA)[0];
            ((uint4*)dstA)[1] = ((const uint4*)srcA)[1];
            ((uint4*)dstB)[0] = ((const uint4*)srcB)[0];
            ((uint4*)dstB)[1] = ((const uint4*)srcB)[1];
#endif
        }
        __syncthreads();

        Frag a[2], b[2];
        for (int mt = 0; mt < 2; ++mt) {
            const unsigned short* p = sA + (waveM + mt * 16 + lrow) * 32;
            a[mt].u[0] = *(const uint4*)(p + kh * 8);        // K = kh*8 .. +7
            a[mt].u[1] = *(const uint4*)(p + 16 + kh * 8);   // K = 16+kh*8 .. +7
        }
        for (int nt = 0; nt < 2; ++nt) {
            const unsigned short* p = sB + (waveN + nt * 16 + lrow) * 32;
            b[nt].u[0] = *(const uint4*)(p + kh * 8);        // K = kh*8 .. +7
            b[nt].u[1] = *(const uint4*)(p + 16 + kh * 8);   // K = 16+kh*8 .. +7
        }
        for (int mt = 0; mt < 2; ++mt)
            for (int nt = 0; nt < 2; ++nt)
                acc[mt][nt] = __builtin_amdgcn_wmma_f32_16x16x32_bf16(
                    false, a[mt].v, false, b[nt].v, (short)0, acc[mt][nt], false, false);
        __syncthreads();
    }

    // poly features -> LDS repack tile (local layout [64 rows][192 cols])
    const float c1 = coeffs[1], c2 = coeffs[2], c3 = coeffs[3];
    for (int mt = 0; mt < 2; ++mt)
        for (int nt = 0; nt < 2; ++nt) {
            int col = waveN + nt * 16 + lrow;          // 0..63 == d within head
            for (int r = 0; r < 8; ++r) {
                int m = waveM + mt * 16 + kh * 8 + r;  // 0..63
                float x  = acc[mt][nt][r];
                float xs = fminf(fmaxf(x, -10.f), 10.f);
                float p1 = fminf(fmaxf(c1 * xs,           -1e6f), 1e6f);
                float p2 = fminf(fmaxf(c2 * xs * xs,      -1e6f), 1e6f);
                float p3 = fminf(fmaxf(c3 * xs * xs * xs, -1e6f), 1e6f);
                sP[m * 192 + col]       = f2bf(p1);
                sP[m * 192 + 64 + col]  = f2bf(p2);
                sP[m * 192 + 128 + col] = f2bf(p3);
            }
        }
    __syncthreads();

    // flush: 2 threads per row, 192B contiguous each -> b128 stores
    {
        int prow = tid >> 1;                 // 0..63
        int seg  = tid & 1;                  // halves of 384B row
        int mg   = bM + prow;
        int head = bN >> 6;                  // one head per N-block
        const uint4* src = (const uint4*)(sP + prow * 192 + seg * 96);
        uint4*       dst = (uint4*)(poly + (mg * 12 + head) * 192 + seg * 96);
        #pragma unroll
        for (int j = 0; j < 12; ++j) dst[j] = src[j];
    }
}

// ------------- fused MLP: H = relu(P@W1nc + bias1); out = H@W2s + b2 -------
// block = 256 threads (8 waves), 32 rows of P per block.
__global__ __launch_bounds__(256) void k_mlp(
    const unsigned short* __restrict__ poly,  // [98304,192] bf16
    const unsigned short* __restrict__ w1t,   // [512,192]   bf16, N-major
    const float* __restrict__ bias1,          // [512] folded bias
    const unsigned short* __restrict__ w2t,   // [64,512]    bf16, N-major
    const float* __restrict__ b2,             // [256], use first 64
    float* __restrict__ out)                  // [8192,768] fp32
{
    __shared__ __align__(16) unsigned short sA[32 * 32];    //  2 KB
    __shared__ __align__(16) unsigned short sH[32 * 512];   // 32 KB hidden tile
    __shared__ __align__(16) float          sO[32 * 64];    //  8 KB output tile

    const int tid  = threadIdx.x;
    const int lane = tid & 31;
    const int wave = tid >> 5;               // 0..7
    const int lrow = lane & 15;
    const int kh   = lane >> 4;
    const int bM   = blockIdx.x * 32;

    // ---- stage B: H = relu(P @ W1nc + bias1) ----
    f32x8 acc[2][4] = {};
    const int n0 = wave * 64;
    for (int kk = 0; kk < 192; kk += 32) {
        {   // stage 32x32 A tile: 8B per thread
            int arow = tid >> 3, q = tid & 7;
            const unsigned short* src = poly + (bM + arow) * 192 + kk + q * 4;
            unsigned short*       dst = sA + arow * 32 + q * 4;
#ifdef USE_ASYNC_LDS
            cp_async_b64(src, dst);
            cp_async_wait();
#else
            *(uint2*)dst = *(const uint2*)src;
#endif
        }
        __syncthreads();
        Frag a[2];
        for (int mt = 0; mt < 2; ++mt) {
            const unsigned short* p = sA + (mt * 16 + lrow) * 32;
            a[mt].u[0] = *(const uint4*)(p + kh * 8);
            a[mt].u[1] = *(const uint4*)(p + 16 + kh * 8);
        }
        for (int nt = 0; nt < 4; ++nt) {
            Frag b;
            const unsigned short* p = w1t + (n0 + nt * 16 + lrow) * 192 + kk;
            b.u[0] = *(const uint4*)(p + kh * 8);
            b.u[1] = *(const uint4*)(p + 16 + kh * 8);
            for (int mt = 0; mt < 2; ++mt)
                acc[mt][nt] = __builtin_amdgcn_wmma_f32_16x16x32_bf16(
                    false, a[mt].v, false, b.v, (short)0, acc[mt][nt], false, false);
        }
        __syncthreads();
    }
    // bias + ReLU -> LDS hidden tile (bf16)
    for (int mt = 0; mt < 2; ++mt)
        for (int nt = 0; nt < 4; ++nt) {
            int n = n0 + nt * 16 + lrow;
            float bia = bias1[n];
            for (int r = 0; r < 8; ++r) {
                int m = mt * 16 + kh * 8 + r;
                sH[m * 512 + n] = f2bf(fmaxf(acc[mt][nt][r] + bia, 0.f));
            }
        }
    __syncthreads();

    // ---- stage C: out = H @ W2s + b2 ; 2x4 tiles, one per wave ----
    const int mt = wave >> 2;     // 0..1
    const int nt = wave & 3;      // 0..3
    f32x8 c = {};
    for (int kk = 0; kk < 512; kk += 32) {
        Frag a, b;
        const unsigned short* pa = sH + (mt * 16 + lrow) * 512 + kk;
        a.u[0] = *(const uint4*)(pa + kh * 8);
        a.u[1] = *(const uint4*)(pa + 16 + kh * 8);
        const unsigned short* pb = w2t + (nt * 16 + lrow) * 512 + kk;
        b.u[0] = *(const uint4*)(pb + kh * 8);
        b.u[1] = *(const uint4*)(pb + 16 + kh * 8);
        c = __builtin_amdgcn_wmma_f32_16x16x32_bf16(
            false, a.v, false, b.v, (short)0, c, false, false);
    }
    {   // bias -> LDS output tile
        int n = nt * 16 + lrow;
        float bia = b2[n];
        for (int r = 0; r < 8; ++r) {
            int m = mt * 16 + kh * 8 + r;
            sO[m * 64 + n] = c[r] + bia;
        }
    }
    __syncthreads();
    {   // flush 32x64 fp32 tile: 32B per thread, rows land at out[rr*768+head*64]
        int orow = tid >> 3;      // 0..31
        int seg  = tid & 7;       // 8 x 8 floats
        int prow = bM + orow;     // = (b*S+s)*12 + head
        int rr   = prow / 12;
        int head = prow - rr * 12;
        const uint4* src = (const uint4*)(sO + orow * 64 + seg * 8);
        uint4*       dst = (uint4*)(out + rr * 768 + head * 64 + seg * 8);
        dst[0] = src[0];
        dst[1] = src[1];
    }
}

// ---------------------------------------------------------------------------

extern "C" void kernel_launch(void* const* d_in, const int* in_sizes, int n_in,
                              void* d_out, int out_size, void* d_ws, size_t ws_size,
                              hipStream_t stream) {
    const float* X      = (const float*)d_in[0];  // [4,2048,768]
    const float* Wq     = (const float*)d_in[1];  // [768,768] (in,out)
    const float* coeffs = (const float*)d_in[2];  // [4]
    const float* W1     = (const float*)d_in[3];  // [256,512]
    const float* b1     = (const float*)d_in[4];  // [512]
    const float* W2     = (const float*)d_in[5];  // [512,256]
    const float* b2     = (const float*)d_in[6];  // [256]
    float* out          = (float*)d_out;          // [4,2048,768]

    char* w = (char*)d_ws;
    unsigned short* Xb    = (unsigned short*)w;  w += 8192ull * 768 * 2;  // 12.6 MB
    unsigned short* Wqt   = (unsigned short*)w;  w += 768ull * 768 * 2;   // 1.18 MB
    unsigned short* W1t   = (unsigned short*)w;  w += 512ull * 192 * 2;   // 0.20 MB
    unsigned short* W2t   = (unsigned short*)w;  w += 64ull * 512 * 2;    // 64 KB
    float*          bias1 = (float*)w;           w += 512 * 4;
    unsigned short* poly  = (unsigned short*)w;  /* 98304*192*2 = 37.7 MB */

    const int NX = 8192 * 768;
    k_cvt_x <<<NX / 1024,     256, 0, stream>>>(X, Xb, NX);
    k_wq_t  <<<589824 / 256,  256, 0, stream>>>(Wq, Wqt);
    k_w1_t  <<<98304 / 256,   256, 0, stream>>>(W1, W1t);
    k_bias1 <<<2,             256, 0, stream>>>(W1, b1, coeffs, bias1);
    k_w2_t  <<<32768 / 256,   256, 0, stream>>>(W2, W2t);

    k_gemm1_poly<<<dim3(8192 / 64, 768 / 64), 128, 0, stream>>>(Xb, Wqt, coeffs, poly);
    k_mlp       <<<98304 / 32,                256, 0, stream>>>(poly, W1t, bias1, W2t, b2, out);
}
